// ReachabilityGNN_24051816857812
// MI455X (gfx1250) — compile-verified, matrix operand order checked by reading, not run on hardware
//
#include <hip/hip_runtime.h>
#include <hip/hip_bf16.h>

typedef __attribute__((ext_vector_type(16))) _Float16 v16h;
typedef __attribute__((ext_vector_type(8)))  _Float16 v8h;
typedef __attribute__((ext_vector_type(8)))  float    v8f;

#define N_NODES 50000
#define N_EDGES 800000
#define N_TOT   (N_EDGES + N_NODES)   // edges + self loops
#define N_GRAPHS 128
#define HID 64
#define NEG_SLOPE 0.2f
#define ALPHA_FLAG 0.05f

// ---------- ordered-uint encoding for float atomic max ----------
__device__ __forceinline__ unsigned fenc(float f) {
    unsigned u = __float_as_uint(f);
    return (u & 0x80000000u) ? ~u : (u | 0x80000000u);
}
__device__ __forceinline__ float fdec(unsigned u) {
    u = (u & 0x80000000u) ? (u & 0x7FFFFFFFu) : ~u;
    return __uint_as_float(u);
}
__device__ __forceinline__ float lrelu(float e) {
    return e > 0.f ? e : NEG_SLOPE * e;
}

// ---------- tiny FiLM MLPs: c = relu(climber@W_c+b_c); gb = c@W_f + b_f ----------
__global__ void k_small(const float* __restrict__ climber,
                        const float* __restrict__ W_c, const float* __restrict__ b_c,
                        const float* __restrict__ W_f1, const float* __restrict__ b_f1,
                        const float* __restrict__ W_f2, const float* __restrict__ b_f2,
                        float* __restrict__ gb1, float* __restrict__ gb2) {
    __shared__ float cs[HID];
    int g = blockIdx.x, t = threadIdx.x;
    if (t < HID) {
        float acc = b_c[t];
        #pragma unroll
        for (int k = 0; k < 4; ++k) acc += climber[g * 4 + k] * W_c[k * HID + t];
        cs[t] = fmaxf(acc, 0.f);
    }
    __syncthreads();
    float a1 = b_f1[t], a2 = b_f2[t];
    #pragma unroll 8
    for (int k = 0; k < HID; ++k) {
        float cv = cs[k];
        a1 += cv * W_f1[k * 128 + t];
        a2 += cv * W_f2[k * 128 + t];
    }
    gb1[g * 128 + t] = a1;
    gb2[g * 128 + t] = a2;
}

// ---------- node init: x = (x_all[:, :6] @ W_in + b_in) * (1+gamma1[b]) + beta1[b] ----------
__global__ void k_node_init(const float* __restrict__ x_all, const int* __restrict__ batch,
                            const float* __restrict__ W_in, const float* __restrict__ b_in,
                            const float* __restrict__ gb1, float* __restrict__ X) {
    int t = blockIdx.x * blockDim.x + threadIdx.x;
    if (t >= N_NODES * HID) return;
    int n = t >> 6, j = t & 63;
    float acc = b_in[j];
    #pragma unroll
    for (int k = 0; k < 6; ++k) acc += x_all[n * 8 + k] * W_in[k * HID + j];
    int g = batch[n];
    X[t] = acc * (1.f + gb1[g * 128 + j]) + gb1[g * 128 + 64 + j];
}

// ---------- WMMA GEMM: Y[N x KOUT] = X[N x 64] @ W[64 x KOUT] (+bias)(relu) ----------
// 256 threads = 8 waves; block covers 128 rows x KOUT cols.
// W is staged once per block into LDS *in WMMA fragment order*
// (WF[tile][frag][lane][0..15]), so each lane pulls its whole 32-byte v16h
// B fragment with two ds_load_b128 instead of 16 scalar ds_load_u16.
// Each wave keeps its two A fragments in VGPRs and sweeps all KOUT/16 column
// tiles, chaining 2 WMMAs per tile -> KOUT/8 v_wmma per wave with full A reuse.
template <int KOUT, bool HAS_BIAS, bool DO_RELU>
__global__ void k_gemm_wmma(const float* __restrict__ X, const float* __restrict__ W,
                            float* __restrict__ Y, int N,
                            const float* __restrict__ bias) {
    constexpr int NT = KOUT / 16;            // column tiles
    __shared__ v16h WF[NT * 2 * 32];         // B fragments, lane-packed (32B each)

    // ---- stage W (f32 -> f16) into fragment layout with 16B LDS stores ----
    // chunk = ((ct*2 + f)*32 + lane)*2 + half ; 8 halves = 8 consecutive k
    for (int chunk = threadIdx.x; chunk < KOUT * 8; chunk += 256) {
        int ct   = chunk >> 7;               // 128 chunks per column tile
        int rem  = chunk & 127;
        int f    = rem >> 6;                 // which of the two K-fragments
        int ln   = (rem >> 1) & 31;          // target lane
        int half = rem & 1;                  // low/high 8 halves of the v16h
        int col  = ct * 16 + (ln & 15);
        int k0   = f * 32 + (ln >> 4) * 8 + half * 16;
        v8h t;
        #pragma unroll
        for (int i = 0; i < 8; ++i) t[i] = (_Float16)W[(k0 + i) * KOUT + col];
        ((v8h*)WF)[chunk] = t;               // ds_store_b128
    }
    __syncthreads();

    const int lane = threadIdx.x & 31;
    const int wave = threadIdx.x >> 5;
    const int hs   = lane >> 4;              // half-wave select
    const int rl   = lane & 15;
    const int kb   = hs * 8;                 // ISA 16-bit fragment K base
    const int rowbase = blockIdx.x * 128 + wave * 16;

    // wave-uniform tail guard: N % 16 == 0, so a wave's 16 rows are all-valid
    // or all-invalid -> single branch, EXEC stays all-ones through the WMMAs.
    if (rowbase >= N) return;

    // ---- A fragments (16 rows x K=64) via global_load_b128 ----
    const int row = rowbase + rl;
    const float4* Xr = (const float4*)(X + row * 64);
    float4 p0 = Xr[(kb >> 2) + 0],  p1 = Xr[(kb >> 2) + 1];     // k = kb..kb+7
    float4 q0 = Xr[(kb >> 2) + 4],  q1 = Xr[(kb >> 2) + 5];     // k = kb+16..kb+23
    float4 r0 = Xr[(kb >> 2) + 8],  r1 = Xr[(kb >> 2) + 9];     // k = kb+32..
    float4 s0 = Xr[(kb >> 2) + 12], s1 = Xr[(kb >> 2) + 13];    // k = kb+48..
    v16h a0, a1;
    a0[0] = (_Float16)p0.x; a0[1] = (_Float16)p0.y; a0[2] = (_Float16)p0.z; a0[3] = (_Float16)p0.w;
    a0[4] = (_Float16)p1.x; a0[5] = (_Float16)p1.y; a0[6] = (_Float16)p1.z; a0[7] = (_Float16)p1.w;
    a0[8] = (_Float16)q0.x; a0[9] = (_Float16)q0.y; a0[10] = (_Float16)q0.z; a0[11] = (_Float16)q0.w;
    a0[12] = (_Float16)q1.x; a0[13] = (_Float16)q1.y; a0[14] = (_Float16)q1.z; a0[15] = (_Float16)q1.w;
    a1[0] = (_Float16)r0.x; a1[1] = (_Float16)r0.y; a1[2] = (_Float16)r0.z; a1[3] = (_Float16)r0.w;
    a1[4] = (_Float16)r1.x; a1[5] = (_Float16)r1.y; a1[6] = (_Float16)r1.z; a1[7] = (_Float16)r1.w;
    a1[8] = (_Float16)s0.x; a1[9] = (_Float16)s0.y; a1[10] = (_Float16)s0.z; a1[11] = (_Float16)s0.w;
    a1[12] = (_Float16)s1.x; a1[13] = (_Float16)s1.y; a1[14] = (_Float16)s1.z; a1[15] = (_Float16)s1.w;

    // ---- sweep all column tiles, reusing A ----
    #pragma unroll
    for (int ct = 0; ct < NT; ++ct) {
        v16h b0 = WF[(ct * 2 + 0) * 32 + lane];   // 2x ds_load_b128
        v16h b1 = WF[(ct * 2 + 1) * 32 + lane];
        v8f c = {};
        c = __builtin_amdgcn_wmma_f32_16x16x32_f16(false, a0, false, b0, (short)0, c, false, false);
        c = __builtin_amdgcn_wmma_f32_16x16x32_f16(false, a1, false, b1, (short)0, c, false, false);

        float bv = 0.f;
        if (HAS_BIAS) bv = bias[ct * 16 + rl];
        #pragma unroll
        for (int r = 0; r < 8; ++r) {
            // D layout: vgpr r -> row r (lanes 0-15) / row r+8 (lanes 16-31)
            float v = HAS_BIAS ? (c[r] + bv) : c[r];
            if (DO_RELU) v = fmaxf(v, 0.f);
            Y[(rowbase + r + hs * 8) * KOUT + ct * 16 + rl] = v;
        }
    }
}

// ---------- attention scores: s_src[n,h] = <xh[n,h,:], a_src[h,:]>, same for dst ----------
__global__ void k_scores(const float* __restrict__ xh,
                         const float* __restrict__ a_src, const float* __restrict__ a_dst,
                         float* __restrict__ ssrc, float* __restrict__ sdst) {
    int t = blockIdx.x * blockDim.x + threadIdx.x;
    if (t >= N_NODES * 2) return;
    int n = t >> 1, h = t & 1;
    float ss = 0.f, sd = 0.f;
    #pragma unroll 8
    for (int c = 0; c < HID; ++c) {
        float v = xh[n * 128 + h * 64 + c];
        ss += v * a_src[h * 64 + c];
        sd += v * a_dst[h * 64 + c];
    }
    ssrc[t] = ss;
    sdst[t] = sd;
}

__global__ void k_clear(unsigned* __restrict__ p, long long n) {
    long long t = (long long)blockIdx.x * blockDim.x + threadIdx.x;
    if (t < n) p[t] = 0u;   // 0u == ordered-encoding of the minimum float, and 0.0f
}

// ---------- edge pass 1: segment max of leaky(e) over dst ----------
__global__ void k_edge_max(const int* __restrict__ ei,
                           const float* __restrict__ ssrc, const float* __restrict__ sdst,
                           unsigned* __restrict__ emax) {
    int t = blockIdx.x * blockDim.x + threadIdx.x;
    if (t >= N_TOT * 2) return;
    int e = t >> 1, h = t & 1;
    int s, d;
    if (e < N_EDGES) {
        __builtin_prefetch(&ei[e + 2048], 0, 1);
        __builtin_prefetch(&ei[N_EDGES + e + 2048], 0, 1);
        s = ei[e]; d = ei[N_EDGES + e];
    } else { s = d = e - N_EDGES; }
    float ev = lrelu(ssrc[s * 2 + h] + sdst[d * 2 + h]);
    atomicMax(&emax[d * 2 + h], fenc(ev));
}

// ---------- edge pass 2: denom[d,h] += exp(e - emax[d,h]) ----------
__global__ void k_edge_denom(const int* __restrict__ ei,
                             const float* __restrict__ ssrc, const float* __restrict__ sdst,
                             const unsigned* __restrict__ emax, float* __restrict__ den) {
    int t = blockIdx.x * blockDim.x + threadIdx.x;
    if (t >= N_TOT * 2) return;
    int e = t >> 1, h = t & 1;
    int s, d;
    if (e < N_EDGES) { s = ei[e]; d = ei[N_EDGES + e]; }
    else             { s = d = e - N_EDGES; }
    float ev = lrelu(ssrc[s * 2 + h] + sdst[d * 2 + h]);
    float p  = __expf(ev - fdec(emax[d * 2 + h]));
    atomicAdd(&den[d * 2 + h], p);
}

// ---------- edge pass 3: acc[d,h,:] += alpha * xh[s,h,:] ----------
__global__ void k_edge_accum(const int* __restrict__ ei,
                             const float* __restrict__ ssrc, const float* __restrict__ sdst,
                             const unsigned* __restrict__ emax, const float* __restrict__ den,
                             const float* __restrict__ xh, float* __restrict__ acc) {
    long long t = (long long)blockIdx.x * blockDim.x + threadIdx.x;
    if (t >= (long long)N_TOT * 128) return;
    int e = (int)(t >> 7);
    int r = (int)(t & 127);
    int h = r >> 6, cc = r & 63;
    int s, d;
    if (e < N_EDGES) { s = ei[e]; d = ei[N_EDGES + e]; }
    else             { s = d = e - N_EDGES; }
    float ev    = lrelu(ssrc[s * 2 + h] + sdst[d * 2 + h]);
    float p     = __expf(ev - fdec(emax[d * 2 + h]));
    float alpha = p / (den[d * 2 + h] + 1e-16f);
    atomicAdd(&acc[d * 128 + h * 64 + cc], alpha * xh[s * 128 + h * 64 + cc]);
}

// ---------- finalize: x = relu(mean_heads(acc) + b); optional FiLM ----------
__global__ void k_finalize(const float* __restrict__ acc, const float* __restrict__ bg,
                           const int* __restrict__ batch, const float* __restrict__ gb,
                           int do_film, float* __restrict__ X) {
    int t = blockIdx.x * blockDim.x + threadIdx.x;
    if (t >= N_NODES * HID) return;
    int n = t >> 6, j = t & 63;
    float v = fmaxf(0.5f * (acc[n * 128 + j] + acc[n * 128 + 64 + j]) + bg[j], 0.f);
    if (do_film) {
        int g = batch[n];
        v = v * (1.f + gb[g * 128 + j]) + gb[g * 128 + 64 + j];
    }
    X[t] = v;
}

// ---------- output head: logits_main + ALPHA_FLAG * logits_flag ----------
__global__ void k_out(const float* __restrict__ hid, const float* __restrict__ x_all,
                      const float* __restrict__ W_cl2, const float* __restrict__ b_cl2,
                      const float* __restrict__ W_fh1, const float* __restrict__ b_fh1,
                      const float* __restrict__ W_fh2, const float* __restrict__ b_fh2,
                      float* __restrict__ out) {
    int t = blockIdx.x * blockDim.x + threadIdx.x;
    if (t >= N_NODES * 4) return;
    int n = t >> 2, o = t & 3;
    float lm = b_cl2[o];
    #pragma unroll 8
    for (int k = 0; k < HID; ++k) lm += hid[n * 64 + k] * W_cl2[k * 4 + o];
    float f0 = x_all[n * 8 + 6], f1 = x_all[n * 8 + 7];
    float lf = b_fh2[o];
    #pragma unroll
    for (int j = 0; j < 8; ++j) {
        float hv = fmaxf(f0 * W_fh1[j] + f1 * W_fh1[8 + j] + b_fh1[j], 0.f);
        lf += hv * W_fh2[j * 4 + o];
    }
    out[t] = lm + ALPHA_FLAG * lf;
}

// ---------------- workspace layout (float units) ----------------
// GB1[16384] GB2[16384] X[3.2M] XH[6.4M] ACC[6.4M] EMAX[100k] DEN[100k] SSRC[100k] SDST[100k]
#define OFF_GB1  0
#define OFF_GB2  16384
#define OFF_X    32768
#define OFF_XH   (OFF_X   + N_NODES * 64)
#define OFF_ACC  (OFF_XH  + N_NODES * 128)
#define OFF_EMAX (OFF_ACC + N_NODES * 128)
#define OFF_DEN  (OFF_EMAX + N_NODES * 2)
#define OFF_SSRC (OFF_DEN  + N_NODES * 2)
#define OFF_SDST (OFF_SSRC + N_NODES * 2)

extern "C" void kernel_launch(void* const* d_in, const int* in_sizes, int n_in,
                              void* d_out, int out_size, void* d_ws, size_t ws_size,
                              hipStream_t stream) {
    const float* x_all  = (const float*)d_in[0];
    const float* climber= (const float*)d_in[1];
    const int*   ei     = (const int*)  d_in[2];
    const int*   batch  = (const int*)  d_in[3];
    const float* W_in   = (const float*)d_in[4];
    const float* b_in   = (const float*)d_in[5];
    const float* W_c    = (const float*)d_in[6];
    const float* b_c    = (const float*)d_in[7];
    const float* W_f1   = (const float*)d_in[8];
    const float* b_f1   = (const float*)d_in[9];
    const float* W_f2   = (const float*)d_in[10];
    const float* b_f2   = (const float*)d_in[11];
    const float* W_g1   = (const float*)d_in[12];
    const float* a_src1 = (const float*)d_in[13];
    const float* a_dst1 = (const float*)d_in[14];
    const float* b_g1   = (const float*)d_in[15];
    const float* W_g2   = (const float*)d_in[16];
    const float* a_src2 = (const float*)d_in[17];
    const float* a_dst2 = (const float*)d_in[18];
    const float* b_g2   = (const float*)d_in[19];
    const float* W_cl1  = (const float*)d_in[20];
    const float* b_cl1  = (const float*)d_in[21];
    const float* W_cl2  = (const float*)d_in[22];
    const float* b_cl2  = (const float*)d_in[23];
    const float* W_fh1  = (const float*)d_in[24];
    const float* b_fh1  = (const float*)d_in[25];
    const float* W_fh2  = (const float*)d_in[26];
    const float* b_fh2  = (const float*)d_in[27];

    float*    ws   = (float*)d_ws;
    float*    GB1  = ws + OFF_GB1;
    float*    GB2  = ws + OFF_GB2;
    float*    X    = ws + OFF_X;
    float*    XH   = ws + OFF_XH;
    float*    ACC  = ws + OFF_ACC;
    unsigned* EMAX = (unsigned*)(ws + OFF_EMAX);
    float*    DEN  = ws + OFF_DEN;
    float*    SSRC = ws + OFF_SSRC;
    float*    SDST = ws + OFF_SDST;
    float*    OUT  = (float*)d_out;

    const int blk = 256;
    // 1) FiLM MLPs
    k_small<<<N_GRAPHS, 128, 0, stream>>>(climber, W_c, b_c, W_f1, b_f1, W_f2, b_f2, GB1, GB2);
    // 2) node embedding + FiLM1
    k_node_init<<<(N_NODES * 64 + blk - 1) / blk, blk, 0, stream>>>(x_all, batch, W_in, b_in, GB1, X);

    const float* Wg[2]  = {W_g1, W_g2};
    const float* As[2]  = {a_src1, a_src2};
    const float* Ad[2]  = {a_dst1, a_dst2};
    const float* Bg[2]  = {b_g1, b_g2};
    const int gemm_blocks = (N_NODES + 127) / 128;   // 8 waves/block, 128 rows/block
    for (int layer = 0; layer < 2; ++layer) {
        // xh = X @ W_g  (WMMA, f16 in / f32 acc, W staged in LDS fragment order)
        k_gemm_wmma<128, false, false><<<gemm_blocks, 256, 0, stream>>>(X, Wg[layer], XH, N_NODES, nullptr);
        // per-head scores
        k_scores<<<(N_NODES * 2 + blk - 1) / blk, blk, 0, stream>>>(XH, As[layer], Ad[layer], SSRC, SDST);
        // clear ACC + EMAX + DEN (contiguous region)
        long long nclr = (long long)N_NODES * 128 + N_NODES * 4;
        k_clear<<<(unsigned)((nclr + blk - 1) / blk), blk, 0, stream>>>((unsigned*)ACC, nclr);
        // three edge passes
        int et2 = N_TOT * 2;
        k_edge_max  <<<(et2 + blk - 1) / blk, blk, 0, stream>>>(ei, SSRC, SDST, EMAX);
        k_edge_denom<<<(et2 + blk - 1) / blk, blk, 0, stream>>>(ei, SSRC, SDST, EMAX, DEN);
        long long ethr = (long long)N_TOT * 128;
        k_edge_accum<<<(unsigned)((ethr + blk - 1) / blk), blk, 0, stream>>>(ei, SSRC, SDST, EMAX, DEN, XH, ACC);
        // mean heads + bias + relu (+FiLM2 after layer 0)
        k_finalize<<<(N_NODES * 64 + blk - 1) / blk, blk, 0, stream>>>(
            ACC, Bg[layer], batch, GB2, layer == 0 ? 1 : 0, X);
    }

    // classifier hidden = relu(X @ W_cl1 + b_cl1)  (WMMA, reuse ACC as hidden buffer)
    k_gemm_wmma<64, true, true><<<gemm_blocks, 256, 0, stream>>>(X, W_cl1, ACC, N_NODES, b_cl1);
    // final logits + flag head
    k_out<<<(N_NODES * 4 + blk - 1) / blk, blk, 0, stream>>>(
        ACC, x_all, W_cl2, b_cl2, W_fh1, b_fh1, W_fh2, b_fh2, OUT);
}
